// NaiveFourierKANLayer_4114578669959
// MI455X (gfx1250) — matrix operand order repaired and checked by reading, not compiled, not run on hardware
//
#include <hip/hip_runtime.h>
#include <type_traits>

// ---- types matching the probe-verified WMMA builtin signatures ----
typedef __attribute__((ext_vector_type(16))) __bf16       v16bf;
typedef __attribute__((ext_vector_type(8)))  __bf16       bf16x8;
typedef __attribute__((ext_vector_type(4)))  __bf16       bf16x4;
typedef __attribute__((ext_vector_type(8)))  float        v8f;
typedef __attribute__((ext_vector_type(4)))  unsigned int u32x4;
typedef __attribute__((ext_vector_type(4)))  float        f32x4;

union Frag { v16bf v; u32x4 u[2]; };

#define GRIDSZ  50
#define IDIM    512
#define ODIM    512
#define MT      32          // M tile per workgroup
#define NT      128         // N tile per workgroup
#define KC      32          // K chunk (one WMMA depth)
#define ASTR    40          // LDS row stride in bf16 (32 + pad), 80B = 16B-aligned
#define BSTR    40
#define KTOT    (2 * GRIDSZ * IDIM)

__global__ __launch_bounds__(256)
void fourier_kan_wmma(const float* __restrict__ x,
                      const float* __restrict__ coef,
                      const float* __restrict__ bias,
                      float* __restrict__ out)
{
    // Ping-pong staged operands (bf16 hi/lo split for bf16x3 precision scheme)
    __shared__ __bf16 sA_hi[2][MT * ASTR];   // A-fragment layout [row][k]
    __shared__ __bf16 sA_lo[2][MT * ASTR];
    __shared__ __bf16 sB_hi[2][NT * BSTR];   // transposed [col][k]
    __shared__ __bf16 sB_lo[2][NT * BSTR];

    const int tid  = threadIdx.x;
    const int lane = tid & 31;
    const int wave = tid >> 5;
    const int wm   = wave & 1;        // 2 waves along M (16 rows each)
    const int wn   = wave >> 1;       // 4 waves along N (32 cols each)
    const int m0   = blockIdx.x * MT; // 2048/32 = 64
    const int n0   = blockIdx.y * NT; // 512/128 = 4

    v8f acc[2] = {};                  // two 16x16 f32 C tiles per wave

    // A-fragment address (16-bit A 16x32 layout):
    // lanes 0-15: M=lane,   K = {0..7, 16..23}
    // lanes16-31: M=lane-16,K = {8..15,24..31}
    const int a_row = wm * 16 + (lane & 15);
    const int khalf = lane >> 4;
    const int aoff  = a_row * ASTR + khalf * 8;

    // A staging: 4 features per thread (32 rows x 32 k = 1024)
    const int fm  = tid >> 3;         // row in tile 0..31
    const int fi0 = (tid & 7) * 4;    // k offset: 0,4,...,28

    int parity = 0;

    auto run_half = [&](int koB, int koE, auto UseCos) {
        constexpr bool UC = decltype(UseCos)::value;

        // Stage one (ko, i0) chunk into buffer p.
        auto stage = [&](int ko, int i0, int p) {
            const float halfk = 0.5f * (float)(ko - koB + 1);
            // ---- A: features = trig(pi*k*x), bf16 hi+lo, packed b64 stores ----
            {
                const float* xp = x + (size_t)(m0 + fm) * IDIM + i0 + fi0;
                f32x4 xq = *(const f32x4*)xp;
                bf16x4 hi, lo;
#pragma unroll
                for (int j = 0; j < 4; ++j) {
                    float rev = xq[j] * halfk;   // v_sin/v_cos take radians/2pi
                    float s;
                    if constexpr (UC) s = __builtin_amdgcn_cosf(rev);
                    else              s = __builtin_amdgcn_sinf(rev);
                    hi[j] = (__bf16)s;
                    lo[j] = (__bf16)(s - (float)hi[j]);
                }
                *(bf16x4*)&sA_hi[p][fm * ASTR + fi0] = hi;  // ds_store_b64
                *(bf16x4*)&sA_lo[p][fm * ASTR + fi0] = lo;
            }
            // ---- B: 32x128 f32 coeff chunk, o-major / k-inner so the LDS
            //      transpose store is one contiguous ds_store_b128 ----
            {
                const size_t kbase = ((size_t)ko * IDIM + i0) * ODIM;
                const bool pf_ok = (ko * IDIM + i0 + KC) < KTOT;
#pragma unroll
                for (int g = 0; g < 2; ++g) {
                    int gi = tid + 256 * g;          // 0..511
                    int o  = gi & 127;               // column in N tile
                    int ig = (gi >> 7) * 8;          // k sub-block: 0,8,16,24
                    const float* cp = coef + kbase + (size_t)ig * ODIM + n0 + o;
                    if (pf_ok)
                        __builtin_prefetch(cp + (size_t)KC * ODIM, 0, 1); // global_prefetch_b8
                    float c[8];
#pragma unroll
                    for (int j = 0; j < 8; ++j)      // coalesced b32 column walk
                        c[j] = cp[(size_t)j * ODIM];
                    bf16x8 hi, lo;
#pragma unroll
                    for (int j = 0; j < 8; ++j) {
                        hi[j] = (__bf16)c[j];
                        lo[j] = (__bf16)(c[j] - (float)hi[j]);
                    }
                    *(bf16x8*)&sB_hi[p][o * BSTR + ig] = hi;  // ds_store_b128
                    *(bf16x8*)&sB_lo[p][o * BSTR + ig] = lo;
                }
            }
        };

        // prologue: stage first chunk of this half
        stage(koB, 0, parity & 1);
        __syncthreads();

        for (int ko = koB; ko < koE; ++ko) {
            for (int i0 = 0; i0 < IDIM; i0 += KC) {
                const int p  = parity & 1;
                const int pn = p ^ 1;
                // stage next chunk (overlaps with WMMA below)
                int ko2 = ko, i02 = i0 + KC;
                if (i02 == IDIM) { ko2 = ko + 1; i02 = 0; }
                if (ko2 < koE) stage(ko2, i02, pn);

                // ---- WMMA: bf16x3 split-precision accumulate ----
                Frag ahi, alo;
                ahi.u[0] = *(const u32x4*)&sA_hi[p][aoff];
                ahi.u[1] = *(const u32x4*)&sA_hi[p][aoff + 16];
                alo.u[0] = *(const u32x4*)&sA_lo[p][aoff];
                alo.u[1] = *(const u32x4*)&sA_lo[p][aoff + 16];
#pragma unroll
                for (int nf = 0; nf < 2; ++nf) {
                    // B 32x16: lanes 0-15 K={0..7,16..23}, lanes 16-31 K={8..15,24..31}, col = lane&15
                    const int boff = (wn * 32 + nf * 16 + (lane & 15)) * BSTR + khalf * 8;
                    Frag bhi, blo;
                    bhi.u[0] = *(const u32x4*)&sB_hi[p][boff];
                    bhi.u[1] = *(const u32x4*)&sB_hi[p][boff + 16];
                    blo.u[0] = *(const u32x4*)&sB_lo[p][boff];
                    blo.u[1] = *(const u32x4*)&sB_lo[p][boff + 16];
                    acc[nf] = __builtin_amdgcn_wmma_f32_16x16x32_bf16(
                        false, ahi.v, false, bhi.v, (short)0, acc[nf], false, false);
                    acc[nf] = __builtin_amdgcn_wmma_f32_16x16x32_bf16(
                        false, ahi.v, false, blo.v, (short)0, acc[nf], false, false);
                    acc[nf] = __builtin_amdgcn_wmma_f32_16x16x32_bf16(
                        false, alo.v, false, bhi.v, (short)0, acc[nf], false, false);
                }
                __syncthreads();   // single barrier per chunk (ping-pong)
                ++parity;
            }
        }
    };

    run_half(0,      GRIDSZ,     std::integral_constant<bool, false>{}); // sin block
    run_half(GRIDSZ, 2 * GRIDSZ, std::integral_constant<bool, true>{});  // cos block

    // ---- epilogue: C layout (VGPR r, lane L): M = 8*(L>=16)+r, N = L&15 ----
    const int n_l = lane & 15;
    const int mh  = lane >> 4;
#pragma unroll
    for (int nf = 0; nf < 2; ++nf) {
        int ng = n0 + wn * 32 + nf * 16 + n_l;
        float b = bias[ng];
#pragma unroll
        for (int r = 0; r < 8; ++r) {
            int mg = m0 + wm * 16 + mh * 8 + r;
            out[(size_t)mg * ODIM + ng] = acc[nf][r] + b;
        }
    }
}

extern "C" void kernel_launch(void* const* d_in, const int* in_sizes, int n_in,
                              void* d_out, int out_size, void* d_ws, size_t ws_size,
                              hipStream_t stream) {
    const float* x    = (const float*)d_in[0]; // (8,256,512)
    const float* coef = (const float*)d_in[1]; // (100,512,512)
    const float* bias = (const float*)d_in[2]; // (512,)
    float*       out  = (float*)d_out;         // (8,256,512)
    (void)in_sizes; (void)n_in; (void)out_size; (void)d_ws; (void)ws_size;

    dim3 grid(2048 / MT, ODIM / NT);           // 64 x 4 = 256 workgroups
    fourier_kan_wmma<<<grid, 256, 0, stream>>>(x, coef, bias, out);
}
